// DeepKoopmanNoDec_48112223650186
// MI455X (gfx1250) — matrix operand order repaired and checked by reading, not compiled, not run on hardware
//
#include <hip/hip_runtime.h>
#include <hip/hip_bf16.h>

typedef __attribute__((ext_vector_type(16))) __bf16 v16bf;
typedef __attribute__((ext_vector_type(8)))  float  v8f;
typedef __attribute__((ext_vector_type(2)))  float  v2f;

#define STATE_DIM  32
#define EMBED_DIM  96
#define LATENT_DIM 128
#define HIDDEN     512
#define BATCH      2048
#define MSTEPS     64
#define EROWS      64     // rows per encoder workgroup
#define NRG        4      // row-groups of waves (16 rows each)
#define NCG        4      // col-groups of waves
#define ETHREADS   (NRG * NCG * 32)   // 512
#define SROWS      16     // rows per scan workgroup
#define APITCH     528    // activation pitch (bf16): 512 + 16 pad, keeps 32B align
#define ZP         132    // scan z-state pitch (f32)

// Permute a column index within its 32-wide K-chunk so that an A-fragment
// (ISA half-wave split K{0-7,16-23 | 8-15,24-31}) is a contiguous 32B run.
__device__ __forceinline__ int kperm(int c) {
  const int g = (c >> 3) & 3;                 // 0,1,2,3 -> 0,2,1,3 (2-bit reverse)
  const int pg = ((g & 1) << 1) | (g >> 1);
  return (c & ~31) | (pg << 3) | (c & 7);
}

// ---------------------------------------------------------------------------
// Weight swizzle: f32 (K x N) -> bf16 fragment-major [kc][tile][lane][e]
// so a B-fragment is one 32-byte contiguous load per lane.
// ---------------------------------------------------------------------------
__global__ __launch_bounds__(256) void swizzle_weights_kernel(
    const float* __restrict__ W, __bf16* __restrict__ out, int K, int N)
{
  const int idx = blockIdx.x * 256 + threadIdx.x;
  if (idx >= K * N) return;
  const int e = idx & 15;
  const int l = (idx >> 4) & 31;
  const int rest = idx >> 9;
  const int nt = N >> 4;
  const int t = rest % nt;
  const int kc = rest / nt;
  const int k = kc * 32 + ((e >> 3) << 4) + ((l >> 4) << 3) + (e & 7);
  const int c = t * 16 + (l & 15);
  out[idx] = (__bf16)W[(size_t)k * N + c];
}

// ---------------------------------------------------------------------------
// One fused MLP layer, k-chunk-outer, software-pipelined:
//   per kc: [load TPW B-frags -> distinct regs] [preload next A-frag]
//           [TPW independent WMMAs]
// so B-loads of later tiles overlap WMMAs of earlier tiles (partial waits).
// ---------------------------------------------------------------------------
template<int K, int N, bool RELU>
__device__ __forceinline__ void mlp_layer_lds(
    const __bf16* hin,                   // LDS, pitch APITCH, K-permuted
    const __bf16* __restrict__ Wf,       // swizzled bf16 weights (global ws)
    const float* __restrict__ bias,
    __bf16* hout,                        // LDS, pitch APITCH, K-permuted
    int lane, int rg, int cg)
{
  constexpr int NT  = N / 16;
  constexpr int NC  = K / 32;
  constexpr int TPW = (NT + NCG - 1) / NCG;
  const int lh = lane >> 4, lm = lane & 15;

  v8f acc[TPW];
#pragma unroll
  for (int i = 0; i < TPW; ++i) acc[i] = {};

  const __bf16* arow = hin + (rg * 16 + lm) * APITCH + lh * 16;
  v16bf a = *(const v16bf*)(arow);
  for (int kc = 0; kc < NC; ++kc) {
    const __bf16* wchunk = Wf + (size_t)kc * NT * 512 + lane * 16;
    v16bf b[TPW];
#pragma unroll
    for (int i = 0; i < TPW; ++i) {
      const int t = cg + NCG * i;
      if ((NT % NCG) && t >= NT) break;          // wave-uniform (cg uniform)
      b[i] = *(const v16bf*)(wchunk + t * 512);
    }
    v16bf an = a;
    if (kc + 1 < NC) {
      an = *(const v16bf*)(arow + (kc + 1) * 32);
      __builtin_prefetch(wchunk + (size_t)NT * 512, 0, 0);
    }
#pragma unroll
    for (int i = 0; i < TPW; ++i) {
      const int t = cg + NCG * i;
      if ((NT % NCG) && t >= NT) break;
      acc[i] = __builtin_amdgcn_wmma_f32_16x16x32_bf16(
          false, a, false, b[i], (short)0, acc[i], false, false);
    }
    a = an;
  }
#pragma unroll
  for (int i = 0; i < TPW; ++i) {
    const int t = cg + NCG * i;
    if ((NT % NCG) && t >= NT) break;
    const int c = t * 16 + lm;
    const float bb = bias[c];
    const int pc = kperm(c);
#pragma unroll
    for (int v = 0; v < 8; ++v) {
      float x = acc[i][v] + bb;
      if (RELU) x = fmaxf(x, 0.0f);
      hout[(rg * 16 + v + 8 * lh) * APITCH + pc] = (__bf16)x;
    }
  }
}

// Final layer: f32 accumulators + bias written straight to global.
template<int K, int N>
__device__ __forceinline__ void mlp_layer_out(
    const __bf16* hin, const __bf16* __restrict__ Wf,
    const float* __restrict__ bias,
    float* __restrict__ gout, int gpitch, int rowbase, int colbase,
    int lane, int rg, int cg)
{
  constexpr int NT  = N / 16;
  constexpr int NC  = K / 32;
  constexpr int TPW = (NT + NCG - 1) / NCG;
  const int lh = lane >> 4, lm = lane & 15;

  v8f acc[TPW];
#pragma unroll
  for (int i = 0; i < TPW; ++i) acc[i] = {};

  const __bf16* arow = hin + (rg * 16 + lm) * APITCH + lh * 16;
  v16bf a = *(const v16bf*)(arow);
  for (int kc = 0; kc < NC; ++kc) {
    const __bf16* wchunk = Wf + (size_t)kc * NT * 512 + lane * 16;
    v16bf b[TPW];
#pragma unroll
    for (int i = 0; i < TPW; ++i) {
      const int t = cg + NCG * i;
      if ((NT % NCG) && t >= NT) break;
      b[i] = *(const v16bf*)(wchunk + t * 512);
    }
    v16bf an = a;
    if (kc + 1 < NC) an = *(const v16bf*)(arow + (kc + 1) * 32);
#pragma unroll
    for (int i = 0; i < TPW; ++i) {
      const int t = cg + NCG * i;
      if ((NT % NCG) && t >= NT) break;
      acc[i] = __builtin_amdgcn_wmma_f32_16x16x32_bf16(
          false, a, false, b[i], (short)0, acc[i], false, false);
    }
    a = an;
  }
#pragma unroll
  for (int i = 0; i < TPW; ++i) {
    const int t = cg + NCG * i;
    if ((NT % NCG) && t >= NT) break;
    const int c = t * 16 + lm;
    const float bb = bias[c];
#pragma unroll
    for (int v = 0; v < 8; ++v)
      gout[(size_t)(rowbase + rg * 16 + v + 8 * lh) * gpitch + colbase + c] =
          acc[i][v] + bb;
  }
}

// ---------------------------------------------------------------------------
// Fused lift: Z[:, 0:32] = X;  Z[:, 32:128] = encoder(X).
// 16 waves = 4 row-groups x 4 col-groups over a 64-row tile; activations stay
// in LDS across all four layers.
// ---------------------------------------------------------------------------
__global__ __launch_bounds__(ETHREADS) void encode_lift_kernel(
    const float* __restrict__ X,
    const __bf16* __restrict__ W1f, const float* __restrict__ b1,
    const __bf16* __restrict__ W2f, const float* __restrict__ b2,
    const __bf16* __restrict__ W3f, const float* __restrict__ b3,
    const __bf16* __restrict__ Wof, const float* __restrict__ bo,
    float* __restrict__ Z)
{
  __shared__ __attribute__((aligned(32))) __bf16 buf0[EROWS * APITCH];
  __shared__ __attribute__((aligned(32))) __bf16 buf1[EROWS * APITCH];
  const int tid  = threadIdx.x;
  const int lane = tid & 31;
  const int wave = tid >> 5;
  const int rg = wave & (NRG - 1), cg = wave >> 2;
  const int rowbase = blockIdx.x * EROWS;

  // Stage input tile (64 x 32) as K-permuted bf16; copy x into Z[:, 0:32].
  for (int i = tid; i < EROWS * STATE_DIM; i += ETHREADS) {  // uniform count
    const int r = i >> 5, c = i & 31;
    const float xv = X[(size_t)(rowbase + r) * STATE_DIM + c];
    buf0[r * APITCH + kperm(c)] = (__bf16)xv;
    Z[(size_t)(rowbase + r) * LATENT_DIM + c] = xv;
  }
  __syncthreads();
  mlp_layer_lds<STATE_DIM, HIDDEN, true>(buf0, W1f, b1, buf1, lane, rg, cg);
  __syncthreads();
  mlp_layer_lds<HIDDEN, HIDDEN, true>(buf1, W2f, b2, buf0, lane, rg, cg);
  __syncthreads();
  mlp_layer_lds<HIDDEN, HIDDEN, true>(buf0, W3f, b3, buf1, lane, rg, cg);
  __syncthreads();
  mlp_layer_out<HIDDEN, EMBED_DIM>(buf1, Wof, bo, Z, LATENT_DIM,
                                   rowbase, STATE_DIM, lane, rg, cg);
}

// ---------------------------------------------------------------------------
// Koopman scan: z_{i+1} = z_i @ A + u_i @ Bmat, 64 serial steps.
// Pure-f32 WMMA (v_wmma_f32_16x16x4_f32); z ping-pongs in LDS; A from L0/L2.
// ---------------------------------------------------------------------------
__global__ __launch_bounds__(256) void koopman_scan_kernel(
    const float* __restrict__ Z0,     // (BATCH, 128)
    const float* __restrict__ U,      // (BATCH, 64, 8)
    const float* __restrict__ Bm,     // (8, 128)
    const float* __restrict__ Akoop,  // (128, 128)
    float* __restrict__ zpred,        // (BATCH, 64, 128)
    float* __restrict__ xpred)        // (BATCH, 64, 32)
{
  __shared__ float zbuf[2][SROWS * ZP];
  __shared__ float bm_lds[8 * LATENT_DIM];
  __shared__ float u_lds[SROWS * 8];
  const int tid  = threadIdx.x;
  const int lane = tid & 31;
  const int wave = tid >> 5;
  const int lh = lane >> 4, lm = lane & 15;
  const int rowbase = blockIdx.x * SROWS;
  const int c0 = wave * 16;

  for (int i = tid; i < 8 * LATENT_DIM; i += 256) bm_lds[i] = Bm[i];
  for (int i = tid; i < SROWS * LATENT_DIM; i += 256) {
    const int r = i >> 7, c = i & 127;
    zbuf[0][r * ZP + c] = Z0[(size_t)(rowbase + r) * LATENT_DIM + c];
  }
  __syncthreads();

  int cur = 0;
  for (int step = 0; step < MSTEPS; ++step) {
    if (tid < SROWS * 8) {
      const int r = tid >> 3, c = tid & 7;
      u_lds[tid] = U[((size_t)(rowbase + r) * MSTEPS + step) * 8 + c];
    }
    __syncthreads();

    v8f acc;   // C init = u_i @ Bmat directly in WMMA C-layout
#pragma unroll
    for (int v = 0; v < 8; ++v) {
      const int r = v + 8 * lh;
      float s = 0.0f;
#pragma unroll
      for (int c = 0; c < 8; ++c)
        s = fmaf(u_lds[r * 8 + c], bm_lds[c * LATENT_DIM + c0 + lm], s);
      acc[v] = s;
    }

    const float* zrow = &zbuf[cur][0];
#pragma unroll 8
    for (int k0 = 0; k0 < LATENT_DIM; k0 += 4) {
      v2f a, b;
      a.x = zrow[lm * ZP + k0 + 2 * lh];
      a.y = zrow[lm * ZP + k0 + 2 * lh + 1];
      b.x = Akoop[(size_t)(k0 + 2 * lh) * LATENT_DIM + c0 + lm];
      b.y = Akoop[(size_t)(k0 + 2 * lh + 1) * LATENT_DIM + c0 + lm];
      acc = __builtin_amdgcn_wmma_f32_16x16x4_f32(
          false, a, false, b, (short)0, acc, false, false);
    }

    const int nxt = cur ^ 1;
#pragma unroll
    for (int v = 0; v < 8; ++v) {
      const int r = v + 8 * lh;
      const float val = acc[v];
      zbuf[nxt][r * ZP + c0 + lm] = val;
      zpred[((size_t)(rowbase + r) * MSTEPS + step) * LATENT_DIM + c0 + lm] = val;
      if (c0 + lm < STATE_DIM)
        xpred[((size_t)(rowbase + r) * MSTEPS + step) * STATE_DIM + c0 + lm] = val;
    }
    __syncthreads();
    cur = nxt;
  }
}

// ---------------------------------------------------------------------------
extern "C" void kernel_launch(void* const* d_in, const int* in_sizes, int n_in,
                              void* d_out, int out_size, void* d_ws, size_t ws_size,
                              hipStream_t stream) {
  const float* x_k    = (const float*)d_in[0];
  const float* u_seq  = (const float*)d_in[1];
  const float* x_next = (const float*)d_in[2];
  const float* W1 = (const float*)d_in[3];
  const float* b1 = (const float*)d_in[4];
  const float* W2 = (const float*)d_in[5];
  const float* b2 = (const float*)d_in[6];
  const float* W3 = (const float*)d_in[7];
  const float* b3 = (const float*)d_in[8];
  const float* Wo = (const float*)d_in[9];
  const float* bo = (const float*)d_in[10];
  const float* Am = (const float*)d_in[11];
  const float* Bm = (const float*)d_in[12];

  float* zpred = (float*)d_out;                                  // (2048,64,128)
  float* xpred = zpred + (size_t)BATCH * MSTEPS * LATENT_DIM;    // (2048,64,32)
  float* ztgt  = xpred + (size_t)BATCH * MSTEPS * STATE_DIM;     // (2048,64,128)

  // Workspace layout
  char* ws = (char*)d_ws;
  float*  z0  = (float*)ws;                                  ws += (size_t)BATCH * LATENT_DIM * 4;
  __bf16* W1f = (__bf16*)ws;                                 ws += (size_t)STATE_DIM * HIDDEN * 2;
  __bf16* W2f = (__bf16*)ws;                                 ws += (size_t)HIDDEN * HIDDEN * 2;
  __bf16* W3f = (__bf16*)ws;                                 ws += (size_t)HIDDEN * HIDDEN * 2;
  __bf16* Wof = (__bf16*)ws;

  // Pre-swizzle weights to bf16 fragment-major (cheap: 0.6M elements).
  swizzle_weights_kernel<<<(STATE_DIM*HIDDEN + 255)/256, 256, 0, stream>>>(W1, W1f, STATE_DIM, HIDDEN);
  swizzle_weights_kernel<<<(HIDDEN*HIDDEN    + 255)/256, 256, 0, stream>>>(W2, W2f, HIDDEN, HIDDEN);
  swizzle_weights_kernel<<<(HIDDEN*HIDDEN    + 255)/256, 256, 0, stream>>>(W3, W3f, HIDDEN, HIDDEN);
  swizzle_weights_kernel<<<(HIDDEN*EMBED_DIM + 255)/256, 256, 0, stream>>>(Wo, Wof, HIDDEN, EMBED_DIM);

  // z_target_seq = lift(x_next): 131072 rows -> 2048 blocks
  encode_lift_kernel<<<(BATCH * MSTEPS) / EROWS, ETHREADS, 0, stream>>>(
      x_next, W1f, b1, W2f, b2, W3f, b3, Wof, bo, ztgt);
  // z_k = lift(x_k): 2048 rows -> 32 blocks
  encode_lift_kernel<<<BATCH / EROWS, ETHREADS, 0, stream>>>(
      x_k, W1f, b1, W2f, b2, W3f, b3, Wof, bo, z0);
  // 64-step recurrence -> z_pred_seq + x_pred_seq
  koopman_scan_kernel<<<BATCH / SROWS, 256, 0, stream>>>(
      z0, u_seq, Bm, Am, zpred, xpred);
}
